// GraphTransformerBlock_37890201486107
// MI455X (gfx1250) — matrix operand order repaired and checked by reading, not compiled, not run on hardware
//
#include <hip/hip_runtime.h>
#include <hip/hip_bf16.h>

typedef __attribute__((ext_vector_type(16))) __bf16 v16bf;
typedef __attribute__((ext_vector_type(8)))  __bf16 v8bf;
typedef __attribute__((ext_vector_type(8)))  float  v8f;

#define D_   512
#define DP_  128
#define H_   8
#define HD_  64
#define B_   4
#define K_   512
#define NT_  (B_ * K_)      // 2048 tokens
#define BH_  (B_ * H_)      // 32 head-batches

// ---------------------------------------------------------------------------
// Weight convert + transpose: src fp32 (R x C) row-major -> dst bf16 (C x R)
// ---------------------------------------------------------------------------
__global__ __launch_bounds__(256)
void conv_transpose_bf16(const float* __restrict__ src, __bf16* __restrict__ dst,
                         int R, int C) {
    size_t i = (size_t)blockIdx.x * blockDim.x + threadIdx.x;
    size_t n = (size_t)R * C;
    if (i >= n) return;
    int r = (int)(i / C);
    int c = (int)(i % C);
    dst[(size_t)c * R + r] = (__bf16)src[i];
}

// ---------------------------------------------------------------------------
// LayerNorm over last dim (512), one wave per row, output bf16
// ---------------------------------------------------------------------------
__global__ __launch_bounds__(256)
void ln_bf16_kernel(const float* __restrict__ x, const float* __restrict__ w,
                    const float* __restrict__ b, __bf16* __restrict__ out, int nrows) {
    int lane = threadIdx.x & 31;
    int wv   = threadIdx.x >> 5;
    int row  = blockIdx.x * 8 + wv;
    if (row >= nrows) return;
    const float* xr = x + (size_t)row * D_;
    float vals[16];
    float s = 0.f, s2 = 0.f;
#pragma unroll
    for (int j = 0; j < 16; ++j) {
        float t = xr[lane + 32 * j];
        vals[j] = t; s += t; s2 += t * t;
    }
#pragma unroll
    for (int off = 16; off; off >>= 1) {
        s  += __shfl_xor(s,  off, 32);
        s2 += __shfl_xor(s2, off, 32);
    }
    float mean = s * (1.f / D_);
    float var  = s2 * (1.f / D_) - mean * mean;
    float rstd = rsqrtf(var + 1e-5f);
    __bf16* orow = out + (size_t)row * D_;
#pragma unroll
    for (int j = 0; j < 16; ++j) {
        int i = lane + 32 * j;
        orow[i] = (__bf16)((vals[j] - mean) * rstd * w[i] + b[i]);
    }
}

// ---------------------------------------------------------------------------
// Edge bias via WMMA: bias[b][h][q][k] = sum_p pairwise[b][q][k][p]*edge_w[p][h]
//                                        + edge_b[h]
// One wave per (b, q, 128-key segment); 8 tiles of 16 keys each.
// A operand = edge_w^T (heads x p), bf16, zero-padded rows 8..15, kept in
// registers for the whole wave. B operand = 16 keys x 32 p of pairwise,
// loaded as contiguous fp32 and converted to bf16 in flight.
// C tile: rows = heads, cols = keys -> coalesced 64B stores per head.
// Writes directly into the scores buffer (scores GEMM accumulates on top).
// ---------------------------------------------------------------------------
__global__ __launch_bounds__(256)
void edge_bias_wmma_kernel(const float* __restrict__ pw, const float* __restrict__ ew,
                           const float* __restrict__ eb, float* __restrict__ scores) {
    const int lane = threadIdx.x & 31;
    const int wv   = threadIdx.x >> 5;
    const unsigned wid = blockIdx.x * 8u + (unsigned)wv;   // 8192 waves
    const unsigned seg = wid & 3u;            // 128-key segment within the row
    const unsigned q   = (wid >> 2) & 511u;
    const unsigned b   = wid >> 11;

    // ---- A operand: edge_w^T (16x32 per step, 4 steps cover p=0..127) ----
    const int head = lane & 15;
    const int koff = (lane >> 4) * 16;
    v16bf a[4];
#pragma unroll
    for (int s = 0; s < 4; ++s) {
#pragma unroll
        for (int j = 0; j < 16; ++j) {
            int p = s * 32 + koff + j;
            float f = (head < H_) ? ew[p * H_ + head] : 0.f;
            a[s][j] = (__bf16)f;
        }
    }

    const float* rowbase = pw + ((size_t)(b * K_ + q) * K_ + seg * 128u) * DP_;
    const int c  = lane & 15;          // key column within tile
    const int kb = (lane >> 4) * 16;   // p-offset half
    float* sbase = scores + ((size_t)(b * H_) * K_ + q) * K_ + seg * 128u;

    for (int t = 0; t < 8; ++t) {
        const float* brow = rowbase + ((size_t)(t * 16 + c)) * DP_ + kb;
        if (t + 2 < 8)
            __builtin_prefetch(rowbase + ((size_t)((t + 2) * 16 + c)) * DP_ + kb, 0, 1);
        v8f acc = {};
#pragma unroll
        for (int s = 0; s < 4; ++s) {
            float4 f0 = *(const float4*)(brow + s * 32 + 0);
            float4 f1 = *(const float4*)(brow + s * 32 + 4);
            float4 f2 = *(const float4*)(brow + s * 32 + 8);
            float4 f3 = *(const float4*)(brow + s * 32 + 12);
            v16bf bb;
            bb[0]  = (__bf16)f0.x; bb[1]  = (__bf16)f0.y;
            bb[2]  = (__bf16)f0.z; bb[3]  = (__bf16)f0.w;
            bb[4]  = (__bf16)f1.x; bb[5]  = (__bf16)f1.y;
            bb[6]  = (__bf16)f1.z; bb[7]  = (__bf16)f1.w;
            bb[8]  = (__bf16)f2.x; bb[9]  = (__bf16)f2.y;
            bb[10] = (__bf16)f2.z; bb[11] = (__bf16)f2.w;
            bb[12] = (__bf16)f3.x; bb[13] = (__bf16)f3.y;
            bb[14] = (__bf16)f3.z; bb[15] = (__bf16)f3.w;
            acc = __builtin_amdgcn_wmma_f32_16x16x32_bf16(false, a[s], false, bb,
                                                          (short)0, acc, false, false);
        }
        // rows 0..7 (= heads) live in lanes 0..15; rows 8..15 are pad
        if (lane < 16) {
#pragma unroll
            for (int r = 0; r < H_; ++r) {
                sbase[((size_t)r * K_) * K_ + t * 16 + c] = acc[r] + eb[r];
            }
        }
    }
}

// ---------------------------------------------------------------------------
// Row softmax over 512, one wave per (b,h,q) row, fp32 in -> bf16 probs
// ---------------------------------------------------------------------------
__global__ __launch_bounds__(256)
void softmax_kernel(const float* __restrict__ s, __bf16* __restrict__ p) {
    int lane = threadIdx.x & 31;
    int wv   = threadIdx.x >> 5;
    size_t row = (size_t)blockIdx.x * 8 + wv;     // 16384 rows
    const float* sr = s + row * K_;
    float v[16];
    float m = -3.4e38f;
#pragma unroll
    for (int j = 0; j < 16; ++j) { v[j] = sr[lane + 32 * j]; m = fmaxf(m, v[j]); }
#pragma unroll
    for (int off = 16; off; off >>= 1) m = fmaxf(m, __shfl_xor(m, off, 32));
    float sum = 0.f;
#pragma unroll
    for (int j = 0; j < 16; ++j) { v[j] = __expf(v[j] - m); sum += v[j]; }
#pragma unroll
    for (int off = 16; off; off >>= 1) sum += __shfl_xor(sum, off, 32);
    float inv = 1.f / sum;
    __bf16* pr = p + row * K_;
#pragma unroll
    for (int j = 0; j < 16; ++j) pr[lane + 32 * j] = (__bf16)(v[j] * inv);
}

// ---------------------------------------------------------------------------
// Generic bf16 WMMA GEMM: one 16x16 C-tile per wave, K-loop of 16x16x32 bf16.
// A: (M x Kd) row-major bf16. B: Bt layout, (N x Kd) row-major bf16.
// ---------------------------------------------------------------------------
struct GemmP {
    const __bf16* A;
    const __bf16* B;
    int lda, ldb, Kd;
    const float* res;    // residual / pre-written bias to accumulate
    const float* bias;   // per-col bias
    float*  outF;
    __bf16* o0;
    __bf16* o1;
    __bf16* o2;
};

// MODE 0: QKV scatter  1: scores(+bias in place)  2: attn*V -> ctx
// MODE 3: proj+residual  4: FFN1+GELU  5: FFN2+residual -> d_out
template <int MODE>
__global__ __launch_bounds__(256)
void gemm_bf16(GemmP p) {
    const int lane = threadIdx.x & 31;
    const int wave = threadIdx.x >> 5;
    const int mt = blockIdx.y * 8 + wave;
    const int nt = blockIdx.x;
    const int z  = blockIdx.z;

    const __bf16* A = p.A;
    const __bf16* B = p.B;
    if (MODE == 1) { A += (size_t)z * K_ * HD_; B += (size_t)z * K_ * HD_; }
    if (MODE == 2) { A += (size_t)z * K_ * K_;  B += (size_t)z * HD_ * K_; }

    const int r16   = lane & 15;
    const int khalf = lane >> 4;
    const __bf16* arow = A + (size_t)(mt * 16 + r16) * p.lda + khalf * 16;
    const __bf16* brow = B + (size_t)(nt * 16 + r16) * p.ldb + khalf * 16;

    v8f acc = {};
    for (int k0 = 0; k0 < p.Kd; k0 += 32) {
        v8bf alo = *(const v8bf*)(arow + k0);
        v8bf ahi = *(const v8bf*)(arow + k0 + 8);
        v16bf a;
#pragma unroll
        for (int i = 0; i < 8; ++i) { a[i] = alo[i]; a[i + 8] = ahi[i]; }
        v16bf b = *(const v16bf*)(brow + k0);
        acc = __builtin_amdgcn_wmma_f32_16x16x32_bf16(false, a, false, b,
                                                      (short)0, acc, false, false);
    }

    const int col   = nt * 16 + (lane & 15);
    const int rbase = mt * 16 + ((lane >> 4) << 3);
#pragma unroll
    for (int r = 0; r < 8; ++r) {
        int row = rbase + r;
        float v = acc[r];
        if (MODE == 0) {
            int t = row, o = col;
            int bq = t >> 9, kk = t & 511;
            int part = o >> 9, oo = o & 511, h = oo >> 6, hd = oo & 63;
            int bh = bq * H_ + h;
            if (part == 0)      p.o0[((size_t)bh * K_ + kk) * HD_ + hd] = (__bf16)(v * 0.125f);
            else if (part == 1) p.o1[((size_t)bh * K_ + kk) * HD_ + hd] = (__bf16)v;
            else                p.o2[((size_t)bh * HD_ + hd) * K_ + kk] = (__bf16)v;
        } else if (MODE == 1) {
            size_t idx = ((size_t)z * K_ + row) * K_ + col;
            p.outF[idx] = p.res[idx] + v;        // res == scores buffer (edge bias)
        } else if (MODE == 2) {
            int bq = z >> 3, h = z & 7;
            p.o0[((size_t)(bq * K_ + row)) * D_ + h * HD_ + col] = (__bf16)v;
        } else if (MODE == 3) {
            size_t idx = (size_t)row * D_ + col;
            p.outF[idx] = p.res[idx] + v + p.bias[col];
        } else if (MODE == 4) {
            float x = v + p.bias[col];
            float g = 0.5f * x * (1.0f + erff(x * 0.70710678118654752f));
            p.o0[(size_t)row * 2048 + col] = (__bf16)g;
        } else { // MODE 5
            size_t idx = (size_t)row * D_ + col;
            p.outF[idx] = p.res[idx] + v + p.bias[col];
        }
    }
}

// ---------------------------------------------------------------------------
extern "C" void kernel_launch(void* const* d_in, const int* in_sizes, int n_in,
                              void* d_out, int out_size, void* d_ws, size_t ws_size,
                              hipStream_t stream) {
    const float* slots  = (const float*)d_in[0];
    const float* pw     = (const float*)d_in[1];
    const float* ln1_w  = (const float*)d_in[2];
    const float* ln1_b  = (const float*)d_in[3];
    const float* ln2_w  = (const float*)d_in[4];
    const float* ln2_b  = (const float*)d_in[5];
    const float* w_qkv  = (const float*)d_in[6];
    const float* edge_w = (const float*)d_in[7];
    const float* edge_b = (const float*)d_in[8];
    const float* out_w  = (const float*)d_in[9];
    const float* out_b  = (const float*)d_in[10];
    const float* ffn_w1 = (const float*)d_in[11];
    const float* ffn_b1 = (const float*)d_in[12];
    const float* ffn_w2 = (const float*)d_in[13];
    const float* ffn_b2 = (const float*)d_in[14];
    float* out = (float*)d_out;

    // ---- carve workspace ----
    char* ws = (char*)d_ws;
    size_t off = 0;
    auto carve = [&](size_t bytes) { void* p = ws + off; off += (bytes + 255) & ~(size_t)255; return p; };
    __bf16* x_bf    = (__bf16*)carve((size_t)NT_ * D_ * 2);        // LN1 output
    __bf16* wqkvT   = (__bf16*)carve((size_t)(3 * D_) * D_ * 2);   // (1536 x 512)
    __bf16* outwT   = (__bf16*)carve((size_t)D_ * D_ * 2);
    __bf16* w1T     = (__bf16*)carve((size_t)(4 * D_) * D_ * 2);   // (2048 x 512)
    __bf16* w2T     = (__bf16*)carve((size_t)D_ * (4 * D_) * 2);   // (512 x 2048)
    __bf16* qbf     = (__bf16*)carve((size_t)BH_ * K_ * HD_ * 2);
    __bf16* kbf     = (__bf16*)carve((size_t)BH_ * K_ * HD_ * 2);
    __bf16* vT      = (__bf16*)carve((size_t)BH_ * HD_ * K_ * 2);
    float*  scores  = (float*) carve((size_t)BH_ * K_ * K_ * 4);   // 32 MB
    __bf16* probs   = (__bf16*)carve((size_t)BH_ * K_ * K_ * 2);   // 16 MB
    __bf16* ctx     = (__bf16*)carve((size_t)NT_ * D_ * 2);
    float*  slots2  = (float*) carve((size_t)NT_ * D_ * 4);
    __bf16* y_bf    = (__bf16*)carve((size_t)NT_ * D_ * 2);
    __bf16* hgelu   = (__bf16*)carve((size_t)NT_ * (4 * D_) * 2);
    (void)ws_size; (void)n_in; (void)in_sizes; (void)out_size;

    // ---- weight prep (fp32 -> bf16, transposed to K-major Bt layout) ----
    {
        size_t n;
        n = (size_t)D_ * 3 * D_;
        conv_transpose_bf16<<<(unsigned)((n + 255) / 256), 256, 0, stream>>>(w_qkv, wqkvT, D_, 3 * D_);
        n = (size_t)D_ * D_;
        conv_transpose_bf16<<<(unsigned)((n + 255) / 256), 256, 0, stream>>>(out_w, outwT, D_, D_);
        n = (size_t)D_ * 4 * D_;
        conv_transpose_bf16<<<(unsigned)((n + 255) / 256), 256, 0, stream>>>(ffn_w1, w1T, D_, 4 * D_);
        n = (size_t)(4 * D_) * D_;
        conv_transpose_bf16<<<(unsigned)((n + 255) / 256), 256, 0, stream>>>(ffn_w2, w2T, 4 * D_, D_);
    }

    // ---- LN1 ----
    ln_bf16_kernel<<<NT_ / 8, 256, 0, stream>>>(slots, ln1_w, ln1_b, x_bf, NT_);

    // ---- QKV GEMM: (2048 x 1536 x 512) with q/k/vT scatter ----
    {
        GemmP p{}; p.A = x_bf; p.B = wqkvT; p.lda = D_; p.ldb = D_; p.Kd = D_;
        p.o0 = qbf; p.o1 = kbf; p.o2 = vT;
        gemm_bf16<0><<<dim3(3 * D_ / 16, NT_ / 128, 1), 256, 0, stream>>>(p);
    }

    // ---- edge bias via WMMA (streams 512 MB of pairwise) -> scores buffer ----
    edge_bias_wmma_kernel<<<(B_ * K_ * 4) / 8, 256, 0, stream>>>(pw, edge_w, edge_b, scores);

    // ---- scores: per-head (512 x 512 x 64), accumulate onto bias ----
    {
        GemmP p{}; p.A = qbf; p.B = kbf; p.lda = HD_; p.ldb = HD_; p.Kd = HD_;
        p.res = scores; p.outF = scores;
        gemm_bf16<1><<<dim3(K_ / 16, K_ / 128, BH_), 256, 0, stream>>>(p);
    }

    // ---- softmax -> bf16 probs ----
    softmax_kernel<<<(BH_ * K_) / 8, 256, 0, stream>>>(scores, probs);

    // ---- attn @ V: per-head (512 x 64 x 512) -> ctx (b,q,D) ----
    {
        GemmP p{}; p.A = probs; p.B = vT; p.lda = K_; p.ldb = K_; p.Kd = K_;
        p.o0 = ctx;
        gemm_bf16<2><<<dim3(HD_ / 16, K_ / 128, BH_), 256, 0, stream>>>(p);
    }

    // ---- out projection + residual: slots2 = slots + ctx @ out_w + out_b ----
    {
        GemmP p{}; p.A = ctx; p.B = outwT; p.lda = D_; p.ldb = D_; p.Kd = D_;
        p.res = slots; p.bias = out_b; p.outF = slots2;
        gemm_bf16<3><<<dim3(D_ / 16, NT_ / 128, 1), 256, 0, stream>>>(p);
    }

    // ---- LN2 ----
    ln_bf16_kernel<<<NT_ / 8, 256, 0, stream>>>(slots2, ln2_w, ln2_b, y_bf, NT_);

    // ---- FFN1 + exact GELU: (2048 x 2048 x 512) -> bf16 ----
    {
        GemmP p{}; p.A = y_bf; p.B = w1T; p.lda = D_; p.ldb = D_; p.Kd = D_;
        p.bias = ffn_b1; p.o0 = hgelu;
        gemm_bf16<4><<<dim3(4 * D_ / 16, NT_ / 128, 1), 256, 0, stream>>>(p);
    }

    // ---- FFN2 + residual: (2048 x 512 x 2048) -> d_out fp32 ----
    {
        GemmP p{}; p.A = hgelu; p.B = w2T; p.lda = 4 * D_; p.ldb = 4 * D_; p.Kd = 4 * D_;
        p.res = slots2; p.bias = ffn_b2; p.outF = out;
        gemm_bf16<5><<<dim3(D_ / 16, NT_ / 128, 1), 256, 0, stream>>>(p);
    }
}
